// NonLocalAggregationModule_88278757802227
// MI455X (gfx1250) — compile-verified, matrix operand order checked by reading, not run on hardware
//
#include <hip/hip_runtime.h>

#define HW     9216   // 96*96
#define SPOOL  2304   // 48*48
#define CIN    128
#define CPROJ  64
#define NFR    4
#define XPAD   136    // padded row length (elems) for LDS transpose tiles

typedef __attribute__((ext_vector_type(16))) __bf16 v16bf;
typedef __attribute__((ext_vector_type(8)))  float  v8f;

// ---------------- helpers ----------------

static __device__ __forceinline__ unsigned short f2bf(float f) {
  union { float f; unsigned u; } x; x.f = f;
  unsigned r = x.u + 0x7fffu + ((x.u >> 16) & 1u);   // round-to-nearest-even
  return (unsigned short)(r >> 16);
}

static __device__ __forceinline__ float bf2f(unsigned short b) {
  return __int_as_float(((int)b) << 16);
}

// Build a v16bf from two contiguous 16-byte chunks (8 bf16 each).
static __device__ __forceinline__ v16bf ld16bf(const unsigned short* p0,
                                               const unsigned short* p1) {
  union { v16bf v; uint4 q[2]; } u;
  u.q[0] = *(const uint4*)p0;
  u.q[1] = *(const uint4*)p1;
  return u.v;
}

static __device__ __forceinline__ v8f vzero() {
  v8f z;
#pragma unroll
  for (int j = 0; j < 8; ++j) z[j] = 0.f;
  return z;
}

static __device__ __forceinline__ v8f wmma_bf16(v16bf a, v16bf b, v8f c) {
  return __builtin_amdgcn_wmma_f32_16x16x32_bf16(false, a, false, b, (short)0, c,
                                                 false, false);
}

// DPP16 lane swaps within the 16-lane half-group (wave32).
template <int CTRL>
static __device__ __forceinline__ float dppmov(float v) {
  int i = __float_as_int(v);
  i = __builtin_amdgcn_update_dpp(i, i, CTRL, 0xF, 0xF, true);
  return __int_as_float(i);
}
static __device__ __forceinline__ float rmax16(float v) {
  v = fmaxf(v, dppmov<0xB1>(v));    // xor1: quad_perm(1,0,3,2)
  v = fmaxf(v, dppmov<0x4E>(v));    // xor2: quad_perm(2,3,0,1)
  v = fmaxf(v, dppmov<0x141>(v));   // xor4: row_half_mirror
  v = fmaxf(v, dppmov<0x140>(v));   // xor8: row_mirror
  return v;
}
static __device__ __forceinline__ float rsum16(float v) {
  v += dppmov<0xB1>(v);
  v += dppmov<0x4E>(v);
  v += dppmov<0x141>(v);
  v += dppmov<0x140>(v);
  return v;
}

// CDNA5 async global->LDS copy, 16B per lane (ASYNCcnt). GVS addressing.
static __device__ __forceinline__ void async_b128(unsigned lds_off, unsigned voff,
                                                  unsigned long long base) {
  asm volatile("global_load_async_to_lds_b128 %0, %1, %2"
               :: "v"(lds_off), "v"(voff), "s"(base) : "memory");
}
static __device__ __forceinline__ void wait_async0() {
  asm volatile("s_wait_asynccnt 0x0" ::: "memory");
}

// ---------------- kernel 1: 1x1 conv projection as WMMA GEMM ----------------
// out[frame][s][ci] = scale * (sum_c W[ci][c]*x[frame][c][s] + b[ci]), bf16.

__global__ __launch_bounds__(128) void proj_kernel(
    const float* __restrict__ x,      // [frames][128][9216]
    const float* __restrict__ W,      // [64][128] (row-major == B N-major layout)
    const float* __restrict__ b,      // [64]
    unsigned short* __restrict__ out, // [frames][9216][64] bf16
    float scale) {
  __shared__ __align__(16) unsigned short xt_lds[64 * XPAD];   // X^T tile [s][c]
  __shared__ __align__(16) unsigned short w_lds[CPROJ * XPAD]; // W [ci][c]
  __shared__ float b_lds[CPROJ];

  const int s0 = blockIdx.x * 64;
  const float* xf = x + (size_t)blockIdx.y * CIN * HW;

  for (int i = threadIdx.x; i < CPROJ * CIN; i += 128) {
    const int ci = i >> 7, c = i & 127;
    w_lds[ci * XPAD + c] = f2bf(W[i]);
  }
  if (threadIdx.x < CPROJ) b_lds[threadIdx.x] = b[threadIdx.x];
  for (int i = threadIdx.x; i < 64 * CIN; i += 128) {
    const int c = i >> 6, sl = i & 63;
    xt_lds[sl * XPAD + c] = f2bf(xf[(size_t)c * HW + s0 + sl]);
  }
  __syncthreads();

  const int wv   = threadIdx.x >> 5;
  const int lane = threadIdx.x & 31;
  const int col  = lane & 15;
  const int kb   = (lane >> 4) << 3;
  const int rb   = (lane >> 4) << 3;
  const int swl  = wv * 16 + col;

  v16bf A[4];
#pragma unroll
  for (int k = 0; k < 4; ++k)
    A[k] = ld16bf(&xt_lds[swl * XPAD + k * 32 + kb],
                  &xt_lds[swl * XPAD + k * 32 + kb + 16]);

  v8f acc[4];
#pragma unroll
  for (int cb = 0; cb < 4; ++cb) {
    acc[cb] = vzero();
#pragma unroll
    for (int k = 0; k < 4; ++k) {
      const v16bf B = ld16bf(&w_lds[(cb * 16 + col) * XPAD + k * 32 + kb],
                             &w_lds[(cb * 16 + col) * XPAD + k * 32 + kb + 16]);
      acc[cb] = wmma_bf16(A[k], B, acc[cb]);
    }
  }

  unsigned short* op = out + (size_t)blockIdx.y * HW * CPROJ;
#pragma unroll
  for (int cb = 0; cb < 4; ++cb) {
    const float bb = b_lds[cb * 16 + col];
#pragma unroll
    for (int j = 0; j < 8; ++j) {
      const int s = s0 + wv * 16 + rb + j;
      op[(size_t)s * CPROJ + cb * 16 + col] = f2bf((acc[cb][j] + bb) * scale);
    }
  }
}

// ---------------- kernel 2: 2x2 maxpool of full-res projections ----------------

__global__ __launch_bounds__(256) void pool_kernel(
    const unsigned short* __restrict__ gfull,  // [n][9216][64]
    const unsigned short* __restrict__ pfull,  // [n][9216][64]
    unsigned short* __restrict__ g,            // [n][64][2304]
    unsigned short* __restrict__ phi) {        // [n][2304][64]
  const int t = blockIdx.x * 256 + threadIdx.x;
  const int n = blockIdx.y;
  const int th = t / 48, tw = t % 48;
  const size_t r0 = ((size_t)((th * 2) * 96 + tw * 2)) * CPROJ;
  const size_t r1 = r0 + CPROJ;
  const size_t r2 = r0 + (size_t)96 * CPROJ;
  const size_t r3 = r2 + CPROJ;
  const unsigned short* gb = gfull + (size_t)n * HW * CPROJ;
  const unsigned short* pb = pfull + (size_t)n * HW * CPROJ;

  union Q { uint4 q; unsigned short s[8]; };
#pragma unroll 1
  for (int c8 = 0; c8 < 8; ++c8) {
    const int co = c8 * 8;
    Q a, b2, c, d, o;
    a.q  = *(const uint4*)(pb + r0 + co);
    b2.q = *(const uint4*)(pb + r1 + co);
    c.q  = *(const uint4*)(pb + r2 + co);
    d.q  = *(const uint4*)(pb + r3 + co);
#pragma unroll
    for (int j = 0; j < 8; ++j)
      o.s[j] = f2bf(fmaxf(fmaxf(bf2f(a.s[j]), bf2f(b2.s[j])),
                          fmaxf(bf2f(c.s[j]), bf2f(d.s[j]))));
    *(uint4*)(phi + ((size_t)n * SPOOL + t) * CPROJ + co) = o.q;
    a.q  = *(const uint4*)(gb + r0 + co);
    b2.q = *(const uint4*)(gb + r1 + co);
    c.q  = *(const uint4*)(gb + r2 + co);
    d.q  = *(const uint4*)(gb + r3 + co);
#pragma unroll
    for (int j = 0; j < 8; ++j) {
      const unsigned short m = f2bf(fmaxf(fmaxf(bf2f(a.s[j]), bf2f(b2.s[j])),
                                          fmaxf(bf2f(c.s[j]), bf2f(d.s[j]))));
      g[((size_t)n * CPROJ + co + j) * SPOOL + t] = m;
    }
  }
}

// ---------------- kernel 3: flash attention (WMMA + double-buffered async LDS) ----------------

__global__ __launch_bounds__(128) void attn_kernel(
    const unsigned short* __restrict__ theta,   // [9216][64] bf16, pre-scaled by 0.125
    const unsigned short* __restrict__ phi,     // [n][2304][64] bf16
    const unsigned short* __restrict__ g,       // [n][64][2304] bf16
    unsigned short* __restrict__ yout) {        // [n][9216][64] bf16
  __shared__ __align__(16) unsigned short phi_lds[2][32][CPROJ];  // 2 x 4KB
  __shared__ __align__(16) unsigned short g_lds[2][CPROJ][32];    // 2 x 4KB
  __shared__ __align__(16) unsigned short p_lds[4][16][32];       // 4KB

  const int n    = blockIdx.y;
  const int wave = threadIdx.x >> 5;
  const int lane = threadIdx.x & 31;
  const int col  = lane & 15;
  const int kb   = (lane >> 4) << 3;
  const int rb   = (lane >> 4) << 3;
  const int s0   = blockIdx.x * 64 + wave * 16;
  const unsigned fid = threadIdx.x;   // 0..127 flat copy index

  const unsigned short* thp = theta + (size_t)(s0 + col) * CPROJ;
  const v16bf A0 = ld16bf(thp + kb,      thp + kb + 16);
  const v16bf A1 = ld16bf(thp + 32 + kb, thp + 32 + kb + 16);

  const unsigned short* phin = phi + (size_t)n * SPOOL * CPROJ;
  const unsigned short* gn   = g   + (size_t)n * CPROJ * SPOOL;
  const unsigned long long phib = (unsigned long long)(uintptr_t)phin;
  const unsigned long long gbse = (unsigned long long)(uintptr_t)gn;
  const unsigned phi_base = (unsigned)(uintptr_t)&phi_lds[0][0][0];
  const unsigned g_base   = (unsigned)(uintptr_t)&g_lds[0][0][0];
  const unsigned gr0 = fid >> 2, gr1 = (fid + 128) >> 2, gq = fid & 3;

  // Issue async HBM->LDS copies for one 32-key chunk into buffer `buf`.
  auto issue = [&](int buf, int t0) {
    const unsigned pdst = phi_base + (unsigned)buf * 4096u + fid * 16;
    const unsigned pv   = (unsigned)(t0 * CPROJ * 2) + fid * 16;
    async_b128(pdst,        pv,        phib);
    async_b128(pdst + 2048, pv + 2048, phib);
    const unsigned gdst = g_base + (unsigned)buf * 4096u + fid * 16;
    async_b128(gdst,        (unsigned)((gr0 * SPOOL + t0 + gq * 8) * 2), gbse);
    async_b128(gdst + 2048, (unsigned)((gr1 * SPOOL + t0 + gq * 8) * 2), gbse);
  };

  v8f acc[4];
  float mrow[8], lrow[8];
#pragma unroll
  for (int k = 0; k < 4; ++k) acc[k] = vzero();
#pragma unroll
  for (int j = 0; j < 8; ++j) { mrow[j] = -1e30f; lrow[j] = 0.f; }

  issue(0, 0);
  int cur = 0;

  for (int t0 = 0; t0 < SPOOL; t0 += 32) {
    wait_async0();       // my copies into buf[cur] complete
    __syncthreads();     // everyone's copies done; prev-iter reads of buf[cur^1] done
    if (t0 + 32 < SPOOL) issue(cur ^ 1, t0 + 32);   // overlap next copy with compute

    const unsigned short (*PH)[CPROJ] = phi_lds[cur];
    const unsigned short (*GL)[32]    = g_lds[cur];

    // -------- QK^T: two 16-col key tiles, K = 64 in two wmmas each --------
    const v16bf B00 = ld16bf(&PH[col][kb],           &PH[col][kb + 16]);
    const v16bf B01 = ld16bf(&PH[col][32 + kb],      &PH[col][32 + kb + 16]);
    const v16bf B10 = ld16bf(&PH[16 + col][kb],      &PH[16 + col][kb + 16]);
    const v16bf B11 = ld16bf(&PH[16 + col][32 + kb], &PH[16 + col][32 + kb + 16]);
    v8f pw0 = wmma_bf16(A1, B01, wmma_bf16(A0, B00, vzero()));
    v8f pw1 = wmma_bf16(A1, B11, wmma_bf16(A0, B10, vzero()));

    // -------- online softmax; DPP16 row reductions --------
    float sc[8], p0a[8], p1a[8];
#pragma unroll
    for (int j = 0; j < 8; ++j) {
      const float cm  = rmax16(fmaxf(pw0[j], pw1[j]));
      const float nmj = fmaxf(mrow[j], cm);
      sc[j]  = __expf(mrow[j] - nmj);
      p0a[j] = __expf(pw0[j] - nmj);
      p1a[j] = __expf(pw1[j] - nmj);
      lrow[j] = lrow[j] * sc[j] + rsum16(p0a[j] + p1a[j]);
      mrow[j] = nmj;
    }
#pragma unroll
    for (int k = 0; k < 4; ++k)
#pragma unroll
      for (int j = 0; j < 8; ++j) acc[k][j] *= sc[j];

    // -------- D-layout -> A-layout transpose of P via per-wave LDS --------
#pragma unroll
    for (int j = 0; j < 8; ++j) {
      p_lds[wave][rb + j][col]      = f2bf(p0a[j]);
      p_lds[wave][rb + j][16 + col] = f2bf(p1a[j]);
    }
    asm volatile("s_wait_dscnt 0x0" ::: "memory");   // DS in-order per wave

    const v16bf PA = ld16bf(&p_lds[wave][col][kb], &p_lds[wave][col][kb + 16]);

    // -------- PV: y(16x64) += P(16x32) @ G(32x64) --------
#pragma unroll
    for (int cb = 0; cb < 4; ++cb) {
      const v16bf GB = ld16bf(&GL[cb * 16 + col][kb], &GL[cb * 16 + col][kb + 16]);
      acc[cb] = wmma_bf16(PA, GB, acc[cb]);
    }
    cur ^= 1;
  }

  // -------- finalize: y = acc / l --------
#pragma unroll
  for (int k = 0; k < 4; ++k)
#pragma unroll
    for (int j = 0; j < 8; ++j) {
      const int s = s0 + rb + j;
      yout[((size_t)n * HW + s) * CPROJ + k * 16 + col] = f2bf(acc[k][j] / lrow[j]);
    }
}

// ---------------- kernel 4: out = curr + y @ Wa^T + ba (WMMA GEMM) ----------------

__global__ __launch_bounds__(128) void outconv_kernel(
    const unsigned short* __restrict__ y,       // [n][9216][64] bf16
    const float* __restrict__ Wa,               // [128][64]
    const float* __restrict__ ba,               // [128]
    const float* __restrict__ curr,             // [128][9216]
    float* __restrict__ out) {                  // [n][128][9216]
  __shared__ __align__(16) unsigned short wa_lds[CIN][CPROJ];
  __shared__ float ba_lds[CIN];

  for (int i = threadIdx.x; i < CIN * CPROJ; i += 128)
    (&wa_lds[0][0])[i] = f2bf(Wa[i]);
  if (threadIdx.x < CIN) ba_lds[threadIdx.x] = ba[threadIdx.x];
  __syncthreads();

  const int n    = blockIdx.y;
  const int wave = threadIdx.x >> 5;
  const int lane = threadIdx.x & 31;
  const int col  = lane & 15;
  const int kb   = (lane >> 4) << 3;
  const int rbs  = (lane >> 4) << 3;
  const int s0   = blockIdx.x * 64 + wave * 16;

  const unsigned short* yr = y + ((size_t)n * HW + s0 + col) * CPROJ;
  const v16bf A0 = ld16bf(yr + kb,      yr + kb + 16);
  const v16bf A1 = ld16bf(yr + 32 + kb, yr + 32 + kb + 16);

  v8f acc[8];
#pragma unroll
  for (int cb = 0; cb < 8; ++cb) {
    const v16bf B0 = ld16bf(&wa_lds[cb * 16 + col][kb],      &wa_lds[cb * 16 + col][kb + 16]);
    const v16bf B1 = ld16bf(&wa_lds[cb * 16 + col][32 + kb], &wa_lds[cb * 16 + col][32 + kb + 16]);
    acc[cb] = wmma_bf16(A1, B1, wmma_bf16(A0, B0, vzero()));
  }

#pragma unroll
  for (int cb = 0; cb < 8; ++cb)
#pragma unroll
    for (int j = 0; j < 8; ++j) {
      const int co = cb * 16 + col;
      const int s  = s0 + rbs + j;
      out[((size_t)n * CIN + co) * HW + s] =
          curr[(size_t)co * HW + s] + acc[cb][j] + ba_lds[co];
    }
}

// ---------------- launch ----------------

extern "C" void kernel_launch(void* const* d_in, const int* in_sizes, int n_in,
                              void* d_out, int out_size, void* d_ws, size_t ws_size,
                              hipStream_t stream) {
  const float* curr = (const float*)d_in[0];
  const float* adja = (const float*)d_in[1];
  const float* Wg   = (const float*)d_in[2];
  const float* bg   = (const float*)d_in[3];
  const float* Wt   = (const float*)d_in[4];
  const float* bt   = (const float*)d_in[5];
  const float* Wp   = (const float*)d_in[6];
  const float* bp   = (const float*)d_in[7];
  const float* Wa   = (const float*)d_in[8];
  const float* ba   = (const float*)d_in[9];
  float* out = (float*)d_out;

  unsigned short* theta = (unsigned short*)d_ws;                  // [9216][64]
  unsigned short* gfull = theta + (size_t)HW * CPROJ;             // [4][9216][64]
  unsigned short* pfull = gfull + (size_t)NFR * HW * CPROJ;       // [4][9216][64]
  unsigned short* phi   = pfull + (size_t)NFR * HW * CPROJ;       // [4][2304][64]
  unsigned short* g     = phi + (size_t)NFR * SPOOL * CPROJ;      // [4][64][2304]
  unsigned short* y     = g + (size_t)NFR * SPOOL * CPROJ;        // [4][9216][64]

  proj_kernel<<<dim3(HW / 64, 1),   128, 0, stream>>>(curr, Wt, bt, theta, 0.125f);
  proj_kernel<<<dim3(HW / 64, NFR), 128, 0, stream>>>(adja, Wg, bg, gfull, 1.f);
  proj_kernel<<<dim3(HW / 64, NFR), 128, 0, stream>>>(adja, Wp, bp, pfull, 1.f);
  pool_kernel<<<dim3(SPOOL / 256, NFR), 256, 0, stream>>>(gfull, pfull, g, phi);
  attn_kernel<<<dim3(HW / 64, NFR), 128, 0, stream>>>(theta, phi, g, y);
  outconv_kernel<<<dim3(HW / 64, NFR), 128, 0, stream>>>(y, Wa, ba, curr, out);
}
